// CrossAttention_7069516169644
// MI455X (gfx1250) — compile-verified, matrix operand order checked by reading, not run on hardware
//
#include <hip/hip_runtime.h>
#include <hip/hip_bf16.h>
#include <math.h>

typedef __attribute__((ext_vector_type(16))) _Float16 v16h;
typedef __attribute__((ext_vector_type(8)))  _Float16 v8h;
typedef __attribute__((ext_vector_type(8)))  float    v8f;

#define BATCH 32
#define NI    1024
#define NT    512
#define DIM   1024
#define SCALE 0.03125f   // DIM^-0.5
#define LNEPS 1e-5f

union V16U { v16h v; v8h h[2]; };

// Load one 16x32 fp16 WMMA operand fragment.
// Source row-major, leading dimension `ld`; row r0+(lane&15) supplies matrix-A
// row m or matrix-B column n (same interleave):
//   lanes 0-15 : K = k0+0..7  and k0+16..23
//   lanes 16-31: K = k0+8..15 and k0+24..31
__device__ __forceinline__ v16h load_frag(const _Float16* __restrict__ src,
                                          int ld, int r0, int lane, int k0) {
  int r  = r0 + (lane & 15);
  int kb = k0 + ((lane >> 4) << 3);
  const _Float16* p = src + (size_t)r * ld + kb;
  V16U u;
  u.h[0] = *(const v8h*)(p);
  u.h[1] = *(const v8h*)(p + 16);
  return u.v;
}

// ---------------------------------------------------------------------------
// LayerNorm: one 256-thread block per row of length DIM.
// Writes fp16 row-major and fp16 transposed so every GEMM operand downstream
// is a contiguous 16B load.
// ---------------------------------------------------------------------------
__global__ void __launch_bounds__(256)
ln_dual_kernel(const float* __restrict__ src,
               const float* __restrict__ gamma,
               const float* __restrict__ beta,
               _Float16* __restrict__ dstRM,
               _Float16* __restrict__ dstT,
               int nRows) {
  const int row = blockIdx.x;          // global row: b*nRows + r
  const int b   = row / nRows;
  const int r   = row - b * nRows;
  const int tid = threadIdx.x;

  const float4 x = ((const float4*)(src + (size_t)row * DIM))[tid];
  float s  = x.x + x.y + x.z + x.w;
  float sq = x.x * x.x + x.y * x.y + x.z * x.z + x.w * x.w;

  __shared__ float2 red[256];
  red[tid] = make_float2(s, sq);
  __syncthreads();
#pragma unroll
  for (int off = 128; off > 0; off >>= 1) {
    if (tid < off) {
      red[tid].x += red[tid + off].x;
      red[tid].y += red[tid + off].y;
    }
    __syncthreads();
  }
  const float mu   = red[0].x * (1.0f / DIM);
  const float var  = red[0].y * (1.0f / DIM) - mu * mu;
  const float rstd = rsqrtf(var + LNEPS);

  const int k0 = tid * 4;
  float xv[4] = {x.x, x.y, x.z, x.w};
  _Float16* rowp = dstRM + (size_t)row * DIM + k0;
  _Float16* tp   = dstT + (size_t)b * DIM * nRows + (size_t)k0 * nRows + r;
#pragma unroll
  for (int e = 0; e < 4; ++e) {
    float v = (xv[e] - mu) * rstd * gamma[k0 + e] + beta[k0 + e];
    rowp[e] = (_Float16)v;
    tp[(size_t)e * nRows] = (_Float16)v;
  }
}

// ---------------------------------------------------------------------------
// Scores + softmax: one 128-thread (4-wave) block per (batch, 16 image rows).
// Wave w computes S[16, w*128 .. w*128+128) as 8 WMMA tiles over K=DIM.
// Tile-granularity software pipeline: while WMMA consumes tile ct's B-frag,
// the load for tile ct+1 (or next chunk's A + B0) is in flight. Live frags:
// a + b + bnext = 24 VGPRs (+64 acc) -> no spills.
// ---------------------------------------------------------------------------
__global__ void __launch_bounds__(128)
scores_softmax_kernel(const _Float16* __restrict__ qn,   // [B,NI,DIM]
                      const _Float16* __restrict__ tn,   // [B,NT,DIM]
                      _Float16* __restrict__ P,          // [B,NI,NT]
                      _Float16* __restrict__ PT) {       // [B,NT,NI]
  const int b    = blockIdx.x >> 6;           // NI/16 = 64 row-blocks
  const int i0   = (blockIdx.x & 63) << 4;
  const int tid  = threadIdx.x;
  const int wave = tid >> 5;
  const int lane = tid & 31;

  const _Float16* qb = qn + (size_t)b * NI * DIM;
  const _Float16* tb = tn + (size_t)b * NT * DIM;
  const int t0 = wave * 128;

  v8f acc[8];
#pragma unroll
  for (int ct = 0; ct < 8; ++ct) acc[ct] = (v8f)(0.0f);

  v16h a  = load_frag(qb, DIM, i0, lane, 0);
  v16h bf = load_frag(tb, DIM, t0, lane, 0);

  for (int k0 = 0; k0 < DIM; k0 += 32) {
#pragma unroll
    for (int ct = 0; ct < 8; ++ct) {
      v16h bn = bf;
      if (ct < 7)
        bn = load_frag(tb, DIM, t0 + (ct + 1) * 16, lane, k0);
      acc[ct] = __builtin_amdgcn_wmma_f32_16x16x32_f16(
          false, a, false, bf, (short)0, acc[ct], false, false);
      if (ct == 7 && k0 + 32 < DIM) {
        a  = load_frag(qb, DIM, i0, lane, k0 + 32);
        bn = load_frag(tb, DIM, t0, lane, k0 + 32);
      }
      bf = bn;
    }
  }

  __shared__ float s_s[16][NT];
  __shared__ float s_red[16][8];
  __shared__ float s_rowmax[16];
  __shared__ float s_rowsum[16];

  const int crow = (lane >> 4) << 3;   // 0 or 8
  const int ccol = lane & 15;
#pragma unroll
  for (int ct = 0; ct < 8; ++ct) {
#pragma unroll
    for (int j = 0; j < 8; ++j)
      s_s[crow + j][t0 + ct * 16 + ccol] = acc[ct][j] * SCALE;
  }
  __syncthreads();

  // Row-wise softmax over NT: 8 threads per row, 64 columns each.
  const int row = tid >> 3;
  const int seg = tid & 7;
  const int cb  = seg * 64;

  float m = -1e30f;
  for (int c = 0; c < 64; ++c) m = fmaxf(m, s_s[row][cb + c]);
  s_red[row][seg] = m;
  __syncthreads();
  if (seg == 0) {
    float mm = s_red[row][0];
#pragma unroll
    for (int q = 1; q < 8; ++q) mm = fmaxf(mm, s_red[row][q]);
    s_rowmax[row] = mm;
  }
  __syncthreads();
  const float mx = s_rowmax[row];
  float ssum = 0.0f;
  for (int c = 0; c < 64; ++c) ssum += __expf(s_s[row][cb + c] - mx);
  s_red[row][seg] = ssum;
  __syncthreads();
  if (seg == 0) {
    float t = 0.0f;
#pragma unroll
    for (int q = 0; q < 8; ++q) t += s_red[row][q];
    s_rowsum[row] = t;
  }
  __syncthreads();
  const float inv = 1.0f / s_rowsum[row];

  _Float16* Prow = P  + ((size_t)b * NI + i0 + row) * NT;
  _Float16* PTb  = PT + (size_t)b * NT * NI;
  for (int c = 0; c < 64; ++c) {
    int t  = cb + c;
    float p = __expf(s_s[row][t] - mx) * inv;
    Prow[t] = (_Float16)p;
    PTb[(size_t)t * NI + i0 + row] = (_Float16)p;
  }
}

// ---------------------------------------------------------------------------
// WMMA GEMM + residual:  out[m,n] = res[m,n] + sum_k A[m,k]*Bt[n,k]
// A: [M,K] fp16 row-major; Bt: [N,K] fp16 row-major (B transposed).
// 128-thread block computes a 128x64 tile: wave w -> 32 rows (2 A-frags) x
// 64 cols (4 B-frags). Tile-granularity pipeline: each B-load overlaps two
// WMMAs. Live frags: a0 + a1 + b + bnext = 32 VGPRs (+64 acc) -> no spills.
// ---------------------------------------------------------------------------
__device__ __forceinline__ void gemm_resid_tile(const _Float16* __restrict__ A,
                                                const _Float16* __restrict__ Bt,
                                                const float* __restrict__ res,
                                                float* __restrict__ out,
                                                int N, int K,
                                                int m0, int n0,
                                                int wave, int lane) {
  const int mrow = m0 + wave * 32;

  // Prefetch the fp32 residual tile (one row per lane covers all 32 rows).
  const float* pre = res + (size_t)(mrow + lane) * N + n0;
  __builtin_prefetch(pre, 0, 3);
  __builtin_prefetch(pre + 32, 0, 3);

  v8f acc[2][4];
#pragma unroll
  for (int rt = 0; rt < 2; ++rt)
#pragma unroll
    for (int ct = 0; ct < 4; ++ct) acc[rt][ct] = (v8f)(0.0f);

  v16h a0 = load_frag(A, K, mrow,      lane, 0);
  v16h a1 = load_frag(A, K, mrow + 16, lane, 0);
  v16h bf = load_frag(Bt, K, n0, lane, 0);

  for (int k0 = 0; k0 < K; k0 += 32) {
#pragma unroll
    for (int ct = 0; ct < 4; ++ct) {
      v16h bn = bf;
      if (ct < 3)
        bn = load_frag(Bt, K, n0 + (ct + 1) * 16, lane, k0);
      acc[0][ct] = __builtin_amdgcn_wmma_f32_16x16x32_f16(
          false, a0, false, bf, (short)0, acc[0][ct], false, false);
      acc[1][ct] = __builtin_amdgcn_wmma_f32_16x16x32_f16(
          false, a1, false, bf, (short)0, acc[1][ct], false, false);
      if (ct == 3 && k0 + 32 < K) {
        a0 = load_frag(A, K, mrow,      lane, k0 + 32);
        a1 = load_frag(A, K, mrow + 16, lane, k0 + 32);
        bn = load_frag(Bt, K, n0, lane, k0 + 32);
      }
      bf = bn;
    }
  }

  const int crow = (lane >> 4) << 3;
  const int ccol = lane & 15;
#pragma unroll
  for (int rt = 0; rt < 2; ++rt)
#pragma unroll
    for (int ct = 0; ct < 4; ++ct)
#pragma unroll
      for (int j = 0; j < 8; ++j) {
        size_t idx = (size_t)(mrow + rt * 16 + crow + j) * N + (n0 + ct * 16 + ccol);
        out[idx] = res[idx] + acc[rt][ct][j];
      }
}

// image_out = img + P @ Kn   (K = NT, B-op from KnT [DIM,NT])
__global__ void __launch_bounds__(128)
attend_image_kernel(const _Float16* __restrict__ P,
                    const _Float16* __restrict__ tnT,
                    const float* __restrict__ img,
                    float* __restrict__ out) {
  const int b  = blockIdx.z;
  const int m0 = blockIdx.y * 128;
  const int n0 = blockIdx.x * 64;
  gemm_resid_tile(P   + (size_t)b * NI * NT,
                  tnT + (size_t)b * DIM * NT,
                  img + (size_t)b * NI * DIM,
                  out + (size_t)b * NI * DIM,
                  DIM, NT, m0, n0, threadIdx.x >> 5, threadIdx.x & 31);
}

// text_out = txt + P^T @ Qn  (K = NI, B-op from QnT [DIM,NI])
__global__ void __launch_bounds__(128)
attend_text_kernel(const _Float16* __restrict__ PT,
                   const _Float16* __restrict__ qnT,
                   const float* __restrict__ txt,
                   float* __restrict__ out) {
  const int b  = blockIdx.z;
  const int m0 = blockIdx.y * 128;
  const int n0 = blockIdx.x * 64;
  gemm_resid_tile(PT  + (size_t)b * NT * NI,
                  qnT + (size_t)b * DIM * NI,
                  txt + (size_t)b * NT * DIM,
                  out + (size_t)b * NT * DIM,
                  DIM, NI, m0, n0, threadIdx.x >> 5, threadIdx.x & 31);
}

extern "C" void kernel_launch(void* const* d_in, const int* in_sizes, int n_in,
                              void* d_out, int out_size, void* d_ws, size_t ws_size,
                              hipStream_t stream) {
  const float* img   = (const float*)d_in[0];   // [32,1024,1024]
  const float* txt   = (const float*)d_in[1];   // [32,512,1024]
  const float* gamma = (const float*)d_in[2];   // [1024]
  const float* beta  = (const float*)d_in[3];   // [1024]

  float* out_img = (float*)d_out;                               // [32,1024,1024]
  float* out_txt = out_img + (size_t)BATCH * NI * DIM;          // [32,512,1024]

  // fp16 workspace layout (~256 MB)
  _Float16* ws  = (_Float16*)d_ws;
  _Float16* qn  = ws;                                    // [B,NI,DIM]
  _Float16* qnT = qn  + (size_t)BATCH * NI * DIM;        // [B,DIM,NI]
  _Float16* tn  = qnT + (size_t)BATCH * NI * DIM;        // [B,NT,DIM]
  _Float16* tnT = tn  + (size_t)BATCH * NT * DIM;        // [B,DIM,NT]
  _Float16* P   = tnT + (size_t)BATCH * NT * DIM;        // [B,NI,NT]
  _Float16* PT  = P   + (size_t)BATCH * NI * NT;         // [B,NT,NI]

  ln_dual_kernel<<<BATCH * NI, 256, 0, stream>>>(img, gamma, beta, qn, qnT, NI);
  ln_dual_kernel<<<BATCH * NT, 256, 0, stream>>>(txt, gamma, beta, tn, tnT, NT);

  scores_softmax_kernel<<<BATCH * (NI / 16), 128, 0, stream>>>(qn, tn, P, PT);

  attend_image_kernel<<<dim3(DIM / 64, NI / 128, BATCH), 128, 0, stream>>>(
      P, tnT, img, out_img);
  attend_text_kernel<<<dim3(DIM / 64, NT / 128, BATCH), 128, 0, stream>>>(
      PT, qnT, txt, out_txt);
}
